// GIN_lrp_x_53283364274285
// MI455X (gfx1250) — matrix operand order repaired and checked by reading, not compiled, not run on hardware
//
#include <hip/hip_runtime.h>
#include <hip/hip_bf16.h>

typedef __attribute__((ext_vector_type(16))) _Float16 v16h;
typedef __attribute__((ext_vector_type(8)))  _Float16 v8h;
typedef __attribute__((ext_vector_type(8)))  float    v8f;

#define GIN_N_NODES 100000
#define GIN_N_EDGES 800000
#define GIN_HID 128
#define GIN_OUT 64

// ---------------- small utility kernels ----------------

__global__ void cvt_f32_f16_kernel(const float* __restrict__ src,
                                   _Float16* __restrict__ dst, int n) {
    int i = blockIdx.x * blockDim.x + threadIdx.x;
    if (i < n) dst[i] = (_Float16)src[i];
}

__global__ void deg_kernel(const int* __restrict__ dst, float* __restrict__ deg,
                           int n_edges) {
    int e = blockIdx.x * blockDim.x + threadIdx.x;
    if (e < n_edges) atomicAdd(&deg[dst[e]], 1.0f);
}

// scatter-add h[src] into agg[dst]; one (edge, feature) pair per thread
__global__ void agg_kernel(const float* __restrict__ h, const int* __restrict__ src,
                           const int* __restrict__ dst, float* __restrict__ agg,
                           int n_edges) {
    int e = blockIdx.x * 2 + (threadIdx.x >> 7);
    int f = threadIdx.x & 127;
    if (e < n_edges) {
        int s = src[e];
        int d = dst[e];
        atomicAdd(&agg[d * GIN_HID + f], h[s * GIN_HID + f]);
    }
}

// agg_io = h + agg_io / max(deg, 1)
__global__ void combine_kernel(const float* __restrict__ h, const float* __restrict__ deg,
                               float* __restrict__ agg_io, int n_nodes) {
    int i = blockIdx.x * blockDim.x + threadIdx.x;
    if (i < n_nodes * GIN_HID) {
        int node = i >> 7;
        float inv = 1.0f / fmaxf(deg[node], 1.0f);
        agg_io[i] = h[i] + agg_io[i] * inv;
    }
}

// ---------------- WMMA GEMM: Out = act(H @ W^T + b) ----------------
// H: [nrows, 128] f32.  Wh: [NCOL, 128] f16 row-major.  Out: [nrows, NCOL] f32.
// Block: 256 threads (8 waves), 64 rows per block.
// Wave w -> row-tile (w/2), col-tiles [(w&1)*TPW, ...), TPW = NCOL/32.

template<int NCOL>
__global__ __launch_bounds__(256)
void gemm_bias_act(const float* __restrict__ H, const _Float16* __restrict__ Wh,
                   const float* __restrict__ bias, float* __restrict__ Out,
                   int nrows, int do_relu) {
    constexpr int PITCH = 136;           // f16 elements per LDS row (16B aligned, deconflicted)
    constexpr int TPW   = NCOL / 32;     // col-tiles per wave
    __shared__ _Float16 Hs[64 * PITCH];
    __shared__ _Float16 Ws[NCOL * PITCH];

    const int tid  = threadIdx.x;
    const int row0 = blockIdx.x * 64;

    // ---- stage activations: 64 rows x 128 f32 -> f16 (2048 float4 loads) ----
    #pragma unroll
    for (int it = 0; it < 8; ++it) {
        int idx = it * 256 + tid;        // quad index; 32 quads per row
        int r   = idx >> 5;
        int c4  = idx & 31;
        float4 v = make_float4(0.f, 0.f, 0.f, 0.f);
        if (row0 + r < nrows) v = ((const float4*)H)[(size_t)(row0 + r) * 32 + c4];
        union { _Float16 h[4]; uint2 u; } pk;
        pk.h[0] = (_Float16)v.x; pk.h[1] = (_Float16)v.y;
        pk.h[2] = (_Float16)v.z; pk.h[3] = (_Float16)v.w;
        *(uint2*)&Hs[r * PITCH + c4 * 4] = pk.u;
    }

    // ---- stage weights (already f16): NCOL*128 f16 = NCOL*16 uint4 ----
    #pragma unroll
    for (int it = 0; it < NCOL / 16; ++it) {
        int idx = it * 256 + tid;        // uint4 index; 16 uint4 per weight row
        int r   = idx >> 4;
        int q   = idx & 15;
        uint4 v = ((const uint4*)Wh)[idx];
        *(uint4*)&Ws[r * PITCH + q * 8] = v;
    }

    __syncthreads();

    const int w       = tid >> 5;
    const int lane    = tid & 31;
    const int half    = lane >> 4;       // which 16-lane half
    const int ln      = lane & 15;
    const int rowtile = w >> 1;          // 0..3
    const int ct0     = (w & 1) * TPW;

    v8f acc[TPW];
    #pragma unroll
    for (int t = 0; t < TPW; ++t) { v8f z = {}; acc[t] = z; }

    #pragma unroll
    for (int k = 0; k < 4; ++k) {        // K = 128 in 4 steps of 32
        const int kb = k * 32;
        // A fragment (16x32 f16): lanes 0-15 row M=ln K=kb+0..7 / kb+16..23,
        // lanes 16-31 row M=ln K=kb+8..15 / kb+24..31  -> two ds_load_b128
        const _Float16* ap = &Hs[(rowtile * 16 + ln) * PITCH + kb + half * 8];
        v8h alo = *(const v8h*)ap;
        v8h ahi = *(const v8h*)(ap + 16);
        v16h a;
        #pragma unroll
        for (int i = 0; i < 8; ++i) { a[i] = alo[i]; a[i + 8] = ahi[i]; }

        #pragma unroll
        for (int t = 0; t < TPW; ++t) {
            // B fragment (32x16 f16): lane column N=ln, K = kb + half*8 + 0..7 / kb + 16 + half*8 + 0..7
            const _Float16* bp = &Ws[((ct0 + t) * 16 + ln) * PITCH + kb + half * 8];
            v8h blo = *(const v8h*)bp;
            v8h bhi = *(const v8h*)(bp + 16);
            v16h b;
            #pragma unroll
            for (int i = 0; i < 8; ++i) { b[i] = blo[i]; b[i + 8] = bhi[i]; }

            acc[t] = __builtin_amdgcn_wmma_f32_16x16x32_f16(
                false, a, false, b, (short)0, acc[t], false, false);
        }
    }

    // ---- epilogue: bias (+ReLU) and store; C/D layout: VGPR r -> M = r + half*8 ----
    #pragma unroll
    for (int t = 0; t < TPW; ++t) {
        const int col = (ct0 + t) * 16 + ln;
        const float bv = bias[col];
        #pragma unroll
        for (int r = 0; r < 8; ++r) {
            int grow = row0 + rowtile * 16 + half * 8 + r;
            if (grow < nrows) {
                float v = acc[t][r] + bv;
                if (do_relu) v = fmaxf(v, 0.0f);
                Out[(size_t)grow * NCOL + col] = v;
            }
        }
    }
}

// ---------------- host-side orchestration ----------------

extern "C" void kernel_launch(void* const* d_in, const int* in_sizes, int n_in,
                              void* d_out, int out_size, void* d_ws, size_t ws_size,
                              hipStream_t stream) {
    const float* x    = (const float*)d_in[0];
    const int*   srcv = (const int*)d_in[1];
    const int*   dstv = (const int*)d_in[2];
    const float* W1   = (const float*)d_in[3];
    const float* b1   = (const float*)d_in[4];
    const float* W2   = (const float*)d_in[5];
    const float* b2   = (const float*)d_in[6];
    const float* W3   = (const float*)d_in[7];
    const float* b3   = (const float*)d_in[8];
    const float* Wo1  = (const float*)d_in[9];
    const float* bo1  = (const float*)d_in[10];
    const float* Wo2  = (const float*)d_in[11];
    const float* bo2  = (const float*)d_in[12];
    float* out = (float*)d_out;

    // workspace layout
    char* ws = (char*)d_ws;
    _Float16* W1h  = (_Float16*)(ws + 0);        // 128*128 f16 = 32768 B
    _Float16* W2h  = (_Float16*)(ws + 32768);
    _Float16* W3h  = (_Float16*)(ws + 65536);
    _Float16* Wo1h = (_Float16*)(ws + 98304);
    _Float16* Wo2h = (_Float16*)(ws + 131072);   // 64*128 f16 = 16384 B
    float* deg  = (float*)(ws + 147456);         // 100000 f32 = 400000 B
    float* bufA = (float*)(ws + 547584);         // 100000*128 f32 = 51.2 MB
    float* bufB = (float*)(ws + 547584 + (size_t)GIN_N_NODES * GIN_HID * 4);

    const size_t featBytes = (size_t)GIN_N_NODES * GIN_HID * sizeof(float);
    const int gblk = (GIN_N_NODES + 63) / 64;
    const int eblk2 = (GIN_N_EDGES + 1) / 2;
    const int cblk = (GIN_N_NODES * GIN_HID + 255) / 256;

    // weights -> f16 (once per launch; deterministic)
    cvt_f32_f16_kernel<<<64, 256, 0, stream>>>(W1,  W1h,  128 * 128);
    cvt_f32_f16_kernel<<<64, 256, 0, stream>>>(W2,  W2h,  128 * 128);
    cvt_f32_f16_kernel<<<64, 256, 0, stream>>>(W3,  W3h,  128 * 128);
    cvt_f32_f16_kernel<<<64, 256, 0, stream>>>(Wo1, Wo1h, 128 * 128);
    cvt_f32_f16_kernel<<<32, 256, 0, stream>>>(Wo2, Wo2h, 64 * 128);

    // degrees
    hipMemsetAsync(deg, 0, GIN_N_NODES * sizeof(float), stream);
    deg_kernel<<<(GIN_N_EDGES + 255) / 256, 256, 0, stream>>>(dstv, deg, GIN_N_EDGES);

    // x2 = fc2(relu(fc1(x)))
    gemm_bias_act<128><<<gblk, 256, 0, stream>>>(x,    W1h, b1, bufA, GIN_N_NODES, 1);
    gemm_bias_act<128><<<gblk, 256, 0, stream>>>(bufA, W2h, b2, bufB, GIN_N_NODES, 0);

    // h = x2 + mean_agg(x2)
    hipMemsetAsync(bufA, 0, featBytes, stream);
    agg_kernel<<<eblk2, 256, 0, stream>>>(bufB, srcv, dstv, bufA, GIN_N_EDGES);
    combine_kernel<<<cblk, 256, 0, stream>>>(bufB, deg, bufA, GIN_N_NODES);

    // h = fc2(relu(fc1(h)))
    gemm_bias_act<128><<<gblk, 256, 0, stream>>>(bufA, W1h, b1, bufB, GIN_N_NODES, 1);
    gemm_bias_act<128><<<gblk, 256, 0, stream>>>(bufB, W2h, b2, bufA, GIN_N_NODES, 0);

    // h = relu(fc3(h))
    gemm_bias_act<128><<<gblk, 256, 0, stream>>>(bufA, W3h, b3, bufB, GIN_N_NODES, 1);

    // h = h + mean_agg(h)
    hipMemsetAsync(bufA, 0, featBytes, stream);
    agg_kernel<<<eblk2, 256, 0, stream>>>(bufB, srcv, dstv, bufA, GIN_N_EDGES);
    combine_kernel<<<cblk, 256, 0, stream>>>(bufB, deg, bufA, GIN_N_NODES);

    // h = relu(fc3(h)); h = relu(fco1(h))
    gemm_bias_act<128><<<gblk, 256, 0, stream>>>(bufA, W3h, b3,  bufB, GIN_N_NODES, 1);
    gemm_bias_act<128><<<gblk, 256, 0, stream>>>(bufB, Wo1h, bo1, bufA, GIN_N_NODES, 1);

    // out = fco2(h)  (64 cols, no relu)
    gemm_bias_act<64><<<gblk, 256, 0, stream>>>(bufA, Wo2h, bo2, out, GIN_N_NODES, 0);
}
